// ConvIntNet_58514634440892
// MI455X (gfx1250) — matrix operand order repaired and checked by reading, compiled, not run on hardware
//
#include <hip/hip_runtime.h>
#include <math.h>

// ---------------------------------------------------------------------------
// ConvIntNet on gfx1250: WMMA f32<-f16 16x16x32 pipeline + async LDS staging.
// ---------------------------------------------------------------------------
#define NCONST 150
#define NF     16
#define NEDGE  22350          // 150*149
#define ETILES 1397           // ceil(22350/16)
#define NB     128

typedef _Float16 half16 __attribute__((ext_vector_type(16)));
typedef float    float8 __attribute__((ext_vector_type(8)));
typedef unsigned int u32x4 __attribute__((ext_vector_type(4)));

union V16H { half16 v; u32x4 q[2]; _Float16 h[16]; };

#if defined(__gfx1250__)
#define HAVE_ASYNC_LDS 1
// Async DMA: global -> LDS, 16B per active lane, tracked by ASYNCcnt.
// VDST = 32-bit LDS byte offset (low 32 bits of a generic LDS pointer),
// VADDR = 64-bit global address pair, no SGPR base ("off").
__device__ __forceinline__ void async_copy_b128(const void* gsrc, void* ldst) {
    unsigned lds_off = (unsigned)(unsigned long long)ldst;
    unsigned long long ga = (unsigned long long)gsrc;
    asm volatile("global_load_async_to_lds_b128 %0, %1, off"
                 :: "v"(lds_off), "v"(ga) : "memory");
}
__device__ __forceinline__ void wait_asynccnt0() {
#if __has_builtin(__builtin_amdgcn_s_wait_asynccnt)
    __builtin_amdgcn_s_wait_asynccnt(0);
#else
    asm volatile("s_wait_asynccnt 0x0" ::: "memory");
#endif
}
#else
#define HAVE_ASYNC_LDS 0
#endif

__device__ __forceinline__ float8 wmma_f16(half16 a, half16 b, float8 c) {
    return __builtin_amdgcn_wmma_f32_16x16x32_f16(false, a, false, b,
                                                  (short)0, c, false, false);
}

// Load one B tile (pre-swizzled into per-lane-contiguous layout by k_pack).
__device__ __forceinline__ half16 load_B(const _Float16* packB, int tile, int lane) {
    V16H u;
    const u32x4* p = (const u32x4*)(packB + tile * 512 + lane * 16);
    u.q[0] = p[0];
    u.q[1] = p[1];
    return u.v;
}

// Build a 16x32 f16 A tile from a row-major LDS tile.
// Lane L holds row M=L&15; g=L>>4 picks the K-halves:
//   halves 0..7  -> K = kbase/2 + 8g + [0..7]   (bytes: m*stride + kbase + 16g)
//   halves 8..15 -> K = kbase/2 + 16 + 8g + ... (bytes: m*stride + kbase + 32 + 16g)
__device__ __forceinline__ half16 loadA_rm(const char* buf, int strideB, int kbase,
                                           int g, int m) {
    V16H u;
    u.q[0] = *(const u32x4*)(buf + m * strideB + kbase + g * 16);
    u.q[1] = *(const u32x4*)(buf + m * strideB + kbase + 32 + g * 16);
    return u.v;
}

// ---------------------------------------------------------------------------
// Kernel 0: pack 6 weight matrices into WMMA B-matrix layout (12 tiles).
// B tile element: lane L -> column N=L&15, half h -> row K=(h&7)+8*(L>>4)+16*(h>>3).
// ---------------------------------------------------------------------------
__global__ __launch_bounds__(512) void k_pack(_Float16* __restrict__ packB,
    const float* __restrict__ eW1, const float* __restrict__ eW2,
    const float* __restrict__ eW3, const float* __restrict__ dW1,
    const float* __restrict__ dW2, const float* __restrict__ dW3) {
    int tile = blockIdx.x;
    int t = threadIdx.x;              // 0..511
    int lane = t >> 4, h = t & 15;
    int K = (h & 7) + 8 * (lane >> 4) + 16 * (h >> 3);
    int N = lane & 15;
    const float* W; int Ks, Ns, k0, n0;
    switch (tile) {
        case 0:  W = eW1; Ks = 32; Ns = 30; k0 = 0;  n0 = 0;  break;
        case 1:  W = eW1; Ks = 32; Ns = 30; k0 = 0;  n0 = 16; break;
        case 2:  W = eW2; Ks = 30; Ns = 15; k0 = 0;  n0 = 0;  break;
        case 3:  W = eW3; Ks = 15; Ns = 6;  k0 = 0;  n0 = 0;  break;
        case 4:  W = dW1; Ks = 22; Ns = 45; k0 = 0;  n0 = 0;  break;
        case 5:  W = dW1; Ks = 22; Ns = 45; k0 = 0;  n0 = 16; break;
        case 6:  W = dW1; Ks = 22; Ns = 45; k0 = 0;  n0 = 32; break;
        case 7:  W = dW2; Ks = 45; Ns = 22; k0 = 0;  n0 = 0;  break;
        case 8:  W = dW2; Ks = 45; Ns = 22; k0 = 0;  n0 = 16; break;
        case 9:  W = dW2; Ks = 45; Ns = 22; k0 = 32; n0 = 0;  break;
        case 10: W = dW2; Ks = 45; Ns = 22; k0 = 32; n0 = 16; break;
        default: W = dW3; Ks = 22; Ns = 6;  k0 = 0;  n0 = 0;  break;
    }
    float v = 0.f;
    if (k0 + K < Ks && n0 + N < Ns) v = W[(k0 + K) * Ns + (n0 + N)];
    packB[tile * 512 + lane * 16 + h] = (_Float16)v;
}

// ---------------------------------------------------------------------------
// Kernel 1: batchnorm (inference) -> f16, zero accumulators.
// ---------------------------------------------------------------------------
__global__ __launch_bounds__(256) void k_bn(const float* __restrict__ x,
    const float* __restrict__ gamma, const float* __restrict__ beta,
    const float* __restrict__ mean,  const float* __restrict__ var,
    _Float16* __restrict__ xh, float* __restrict__ effred,
    float* __restrict__ pooled) {
    int idx = blockIdx.x * 256 + threadIdx.x;    // grid=1200 -> 307200 threads
    if (idx < NB * NCONST * NF) {
        int f = idx & 15;
        float sc = gamma[f] * rsqrtf(var[f] + 1e-3f);
        float sh = beta[f] - mean[f] * sc;
        xh[idx] = (_Float16)(x[idx] * sc + sh);
    }
    if (idx < NB * NCONST * 8) effred[idx] = 0.f;
    if (idx < NB * 8) pooled[idx] = 0.f;
}

// ---------------------------------------------------------------------------
// Kernel 2 (hot): per-edge effects MLP 32->30->15->6 via WMMA, scatter-add
// to receiver nodes (edges sorted by receiver: 149 consecutive per node).
// grid (128 batches, 8 chunks) x 256 threads (8 waves).
// ---------------------------------------------------------------------------
__global__ __launch_bounds__(256) void k_effects(
    const _Float16* __restrict__ xh, const _Float16* __restrict__ packB,
    const float* __restrict__ eb1, const float* __restrict__ eb2,
    const float* __restrict__ eb3, float* __restrict__ effred) {
    __shared__ char smem[4800 + 8 * 2048];
    const int b = blockIdx.x;
    const int tid = threadIdx.x;
    const int lane = tid & 31, wave = tid >> 5;
    const int g = lane >> 4, n = lane & 15;

    // stage xh[b] (150x16 f16 = 4800B = 300 x b128) into LDS
    const u32x4* src = (const u32x4*)(xh + (size_t)b * NCONST * NF);
    u32x4* dst = (u32x4*)smem;
#if HAVE_ASYNC_LDS
    for (int i = tid; i < 300; i += 256)
        async_copy_b128(src + i, dst + i);
    wait_asynccnt0();            // my async loads landed in LDS
#else
    __builtin_prefetch(src, 0, 0);               // global_prefetch_b8
    for (int i = tid; i < 300; i += 256) dst[i] = src[i];
#endif
    __syncthreads();             // everyone's loads landed

    const half16 B1a = load_B(packB, 0, lane);
    const half16 B1b = load_B(packB, 1, lane);
    const half16 B2  = load_B(packB, 2, lane);
    const half16 B3  = load_B(packB, 3, lane);

    const float bias1a = eb1[n];
    const float bias1b = (n + 16 < 30) ? eb1[n + 16] : 0.f;
    const float bias2  = (n < 15) ? eb2[n] : 0.f;
    const float bias3  = (n < 6)  ? eb3[n] : 0.f;

    char* bufh1 = smem + 4800 + wave * 2048;     // 16x32 f16, 64B rows
    char* bufh2 = bufh1 + 1024;                  // 16x32 f16, 64B rows
    {   // zero wave-private bounce buffers once (pads stay zero)
        u32x4 z = (u32x4)0u;
        u32x4* p = (u32x4*)bufh1;
        #pragma unroll
        for (int i = 0; i < 4; i++) p[lane + 32 * i] = z;
    }

    for (int t = blockIdx.y * 8 + wave; t < ETILES; t += 64) {
        // ---- build A: 16 edges x 32 features (recv | send) ----
        int e  = t * 16 + n;
        int ec = e < NEDGE ? e : NEDGE - 1;
        int r  = ec / 149;
        int rm = ec - r * 149;
        int s  = rm + (rm >= r ? 1 : 0);
        V16H ua;
        ua.q[0] = *(const u32x4*)(smem + r * 32 + g * 16);  // feats 8g..8g+7 of r
        ua.q[1] = *(const u32x4*)(smem + s * 32 + g * 16);  // feats 8g..8g+7 of s

        // ---- layer 1: 32 -> 30 (2 N-tiles) ----
        float8 d0 = {}, d1 = {};
        d0 = wmma_f16(ua.v, B1a, d0);
        d1 = wmma_f16(ua.v, B1b, d1);
        _Float16* h1 = (_Float16*)bufh1;
        #pragma unroll
        for (int i = 0; i < 8; i++) {
            int m = i + 8 * g;
            h1[m * 32 + n]      = (_Float16)fmaxf(d0[i] + bias1a, 0.f);
            h1[m * 32 + n + 16] = (_Float16)fmaxf(d1[i] + bias1b, 0.f); // cols>=30 -> 0
        }

        // ---- layer 2: 30 -> 15 ----
        half16 a2 = loadA_rm(bufh1, 64, 0, g, n);
        float8 d2 = {};
        d2 = wmma_f16(a2, B2, d2);
        _Float16* h2 = (_Float16*)bufh2;
        #pragma unroll
        for (int i = 0; i < 8; i++) {
            int m = i + 8 * g;
            h2[m * 32 + n] = (_Float16)fmaxf(d2[i] + bias2, 0.f);  // col 15 -> 0
        }

        // ---- layer 3: 15 -> 6 ----
        half16 a3 = loadA_rm(bufh2, 64, 0, g, n);
        float8 d3 = {};
        d3 = wmma_f16(a3, B3, d3);

        // ---- segmented scatter-add to receiver nodes ----
        if (n < 6) {
            float acc = 0.f;
            int curR = -1;
            #pragma unroll
            for (int i = 0; i < 8; i++) {
                int m  = i + 8 * g;
                int ee = t * 16 + m;
                if (ee < NEDGE) {
                    int rr = ee / 149;
                    float v = fmaxf(d3[i] + bias3, 0.f);
                    if (rr != curR) {
                        if (curR >= 0)
                            atomicAdd(&effred[((size_t)b * NCONST + curR) * 8 + n], acc);
                        curR = rr; acc = 0.f;
                    }
                    acc += v;
                }
            }
            if (curR >= 0)
                atomicAdd(&effred[((size_t)b * NCONST + curR) * 8 + n], acc);
        }
    }
}

// ---------------------------------------------------------------------------
// Kernel 3: per-node dynamics MLP 22->45->22->6 via WMMA + pooled sum.
// grid 128, block 320 (10 waves = 10 node tiles of 16).
// ---------------------------------------------------------------------------
__global__ __launch_bounds__(320) void k_dynamics(
    const _Float16* __restrict__ xh, const float* __restrict__ effred,
    const _Float16* __restrict__ packB,
    const float* __restrict__ db1, const float* __restrict__ db2,
    const float* __restrict__ db3, float* __restrict__ pooled) {
    __shared__ char smem[9632 + 10 * 3072];
    const int b = blockIdx.x;
    const int tid = threadIdx.x;
    const int lane = tid & 31, wave = tid >> 5;
    const int g = lane >> 4, n = lane & 15;

    // stage xh[b] (4800B) and effred[b] (150x8 f32 = 4800B)
    const u32x4* sx = (const u32x4*)(xh + (size_t)b * NCONST * NF);
    const u32x4* se = (const u32x4*)(effred + (size_t)b * NCONST * 8);
    u32x4* dx = (u32x4*)smem;
    u32x4* de = (u32x4*)(smem + 4800);
#if HAVE_ASYNC_LDS
    for (int i = tid; i < 300; i += 320) {
        async_copy_b128(sx + i, dx + i);
        async_copy_b128(se + i, de + i);
    }
    wait_asynccnt0();
#else
    for (int i = tid; i < 300; i += 320) { dx[i] = sx[i]; de[i] = se[i]; }
#endif
    float* pool = (float*)(smem + 9600);
    if (tid < 8) pool[tid] = 0.f;
    __syncthreads();

    half16 B1[3], B2[2][2], B3;
    #pragma unroll
    for (int j = 0; j < 3; j++) B1[j] = load_B(packB, 4 + j, lane);
    #pragma unroll
    for (int ks = 0; ks < 2; ks++)
        #pragma unroll
        for (int nt = 0; nt < 2; nt++) B2[ks][nt] = load_B(packB, 7 + ks * 2 + nt, lane);
    B3 = load_B(packB, 11, lane);

    float bias1[3], bias2[2];
    #pragma unroll
    for (int j = 0; j < 3; j++) { int nn = n + 16 * j; bias1[j] = (nn < 45) ? db1[nn] : 0.f; }
    #pragma unroll
    for (int j = 0; j < 2; j++) { int nn = n + 16 * j; bias2[j] = (nn < 22) ? db2[nn] : 0.f; }
    const float bias3 = (n < 6) ? db3[n] : 0.f;

    char* bufh1 = smem + 9632 + wave * 3072;     // 16x64 f16, 128B rows
    char* bufh2 = bufh1 + 2048;                  // 16x32 f16, 64B rows
    {   // zero h1 buf (cols 48..63 must stay zero)
        u32x4 z = (u32x4)0u;
        u32x4* p = (u32x4*)bufh1;
        #pragma unroll
        for (int i = 0; i < 4; i++) p[lane + 32 * i] = z;
    }

    const int base  = wave * 16;                 // node tile base (0..144)
    const int node  = base + n;
    const int nodeC = node < NCONST ? node : NCONST - 1;

    // ---- build A: 16 nodes x 32 (16 feats | 6 effects | pad) ----
    V16H ua;
    ua.q[0] = *(const u32x4*)(smem + nodeC * 32 + g * 16);
    const float* er = (const float*)(smem + 4800 + nodeC * 32);
    const bool lo = (g == 0);
    ua.h[8]  = (_Float16)(lo ? er[0] : 0.f);
    ua.h[9]  = (_Float16)(lo ? er[1] : 0.f);
    ua.h[10] = (_Float16)(lo ? er[2] : 0.f);
    ua.h[11] = (_Float16)(lo ? er[3] : 0.f);
    ua.h[12] = (_Float16)(lo ? er[4] : 0.f);
    ua.h[13] = (_Float16)(lo ? er[5] : 0.f);
    ua.h[14] = (_Float16)0.f;
    ua.h[15] = (_Float16)0.f;

    // ---- layer 1: 22 -> 45 (3 N-tiles) ----
    _Float16* h1 = (_Float16*)bufh1;
    #pragma unroll
    for (int j = 0; j < 3; j++) {
        float8 d = {};
        d = wmma_f16(ua.v, B1[j], d);
        #pragma unroll
        for (int i = 0; i < 8; i++) {
            int m = i + 8 * g;
            h1[m * 64 + n + 16 * j] = (_Float16)fmaxf(d[i] + bias1[j], 0.f);
        }
    }

    // ---- layer 2: 45 -> 22 (2 K-steps x 2 N-tiles) ----
    half16 a20 = loadA_rm(bufh1, 128, 0, g, n);
    half16 a21 = loadA_rm(bufh1, 128, 64, g, n);
    _Float16* h2 = (_Float16*)bufh2;
    #pragma unroll
    for (int nt = 0; nt < 2; nt++) {
        float8 d = {};
        d = wmma_f16(a20, B2[0][nt], d);
        d = wmma_f16(a21, B2[1][nt], d);
        #pragma unroll
        for (int i = 0; i < 8; i++) {
            int m = i + 8 * g;
            h2[m * 32 + n + 16 * nt] = (_Float16)fmaxf(d[i] + bias2[nt], 0.f);
        }
    }

    // ---- layer 3: 22 -> 6 + pooled reduction ----
    half16 a3 = loadA_rm(bufh2, 64, 0, g, n);
    float8 d3 = {};
    d3 = wmma_f16(a3, B3, d3);
    if (n < 6) {
        float acc = 0.f;
        #pragma unroll
        for (int i = 0; i < 8; i++) {
            int m = i + 8 * g;
            if (base + m < NCONST) acc += fmaxf(d3[i] + bias3, 0.f);
        }
        atomicAdd(&pool[n], acc);                // ds_add f32
    }
    __syncthreads();
    if (tid < 6) pooled[b * 8 + tid] = pool[tid];
}

// ---------------------------------------------------------------------------
// Kernel 4: abstract MLP 6 -> 48 -> 5 + softmax. One thread per batch row.
// ---------------------------------------------------------------------------
__global__ __launch_bounds__(128) void k_abstract(const float* __restrict__ pooled,
    const float* __restrict__ aW1, const float* __restrict__ ab1,
    const float* __restrict__ aW2, const float* __restrict__ ab2,
    float* __restrict__ out) {
    int b = threadIdx.x;
    float p[6];
    #pragma unroll
    for (int c = 0; c < 6; c++) p[c] = pooled[b * 8 + c];
    float logits[5];
    #pragma unroll
    for (int k = 0; k < 5; k++) logits[k] = ab2[k];
    for (int j = 0; j < 48; j++) {
        float h = ab1[j];
        #pragma unroll
        for (int c = 0; c < 6; c++) h += p[c] * aW1[c * 48 + j];
        h = fmaxf(h, 0.f);
        #pragma unroll
        for (int k = 0; k < 5; k++) logits[k] += h * aW2[j * 5 + k];
    }
    float mx = logits[0];
    #pragma unroll
    for (int k = 1; k < 5; k++) mx = fmaxf(mx, logits[k]);
    float sum = 0.f, e[5];
    #pragma unroll
    for (int k = 0; k < 5; k++) { e[k] = expf(logits[k] - mx); sum += e[k]; }
    float inv = 1.f / sum;
    #pragma unroll
    for (int k = 0; k < 5; k++) out[b * 5 + k] = e[k] * inv;
}

// ---------------------------------------------------------------------------
extern "C" void kernel_launch(void* const* d_in, const int* in_sizes, int n_in,
                              void* d_out, int out_size, void* d_ws, size_t ws_size,
                              hipStream_t stream) {
    const float* x        = (const float*)d_in[0];
    const float* bn_gamma = (const float*)d_in[3];
    const float* bn_beta  = (const float*)d_in[4];
    const float* bn_mean  = (const float*)d_in[5];
    const float* bn_var   = (const float*)d_in[6];
    const float* eW1 = (const float*)d_in[7];  const float* eb1 = (const float*)d_in[8];
    const float* eW2 = (const float*)d_in[9];  const float* eb2 = (const float*)d_in[10];
    const float* eW3 = (const float*)d_in[11]; const float* eb3 = (const float*)d_in[12];
    const float* dW1 = (const float*)d_in[13]; const float* db1 = (const float*)d_in[14];
    const float* dW2 = (const float*)d_in[15]; const float* db2 = (const float*)d_in[16];
    const float* dW3 = (const float*)d_in[17]; const float* db3 = (const float*)d_in[18];
    const float* aW1 = (const float*)d_in[19]; const float* ab1 = (const float*)d_in[20];
    const float* aW2 = (const float*)d_in[21]; const float* ab2 = (const float*)d_in[22];

    char* ws = (char*)d_ws;
    _Float16* packB  = (_Float16*)ws;                 //     12,288 B (12 B-tiles)
    _Float16* xh     = (_Float16*)(ws + 12288);       //    614,400 B (B,150,16) f16
    float*    effred = (float*)(ws + 626688);         //    614,400 B (B,150,8) f32
    float*    pooled = (float*)(ws + 1241088);        //      4,096 B (B,8) f32

    k_pack<<<12, 512, 0, stream>>>(packB, eW1, eW2, eW3, dW1, dW2, dW3);
    k_bn<<<1200, 256, 0, stream>>>(x, bn_gamma, bn_beta, bn_mean, bn_var,
                                   xh, effred, pooled);
    k_effects<<<dim3(NB, 8), 256, 0, stream>>>(xh, packB, eb1, eb2, eb3, effred);
    k_dynamics<<<NB, 320, 0, stream>>>(xh, effred, packB, db1, db2, db3, pooled);
    k_abstract<<<1, 128, 0, stream>>>(pooled, aW1, ab1, aW2, ab2, (float*)d_out);
}
